// ATTEfficient_25975962206537
// MI455X (gfx1250) — matrix operand order repaired and checked by reading, not compiled, not run on hardware
//
#include <hip/hip_runtime.h>
#include <hip/hip_bf16.h>
#include <math.h>

#define TOTAL  32768
#define N_SEG  128
#define D      1280
#define D_HEAD 512

typedef __attribute__((ext_vector_type(16))) __bf16 v16bf;
typedef __attribute__((ext_vector_type(8)))  float  v8f;

union FragBF { uint4 u[2]; v16bf v; };

__device__ __forceinline__ unsigned short f2bf(float x) {
    unsigned u = __float_as_uint(x);
    u += 0x7FFFu + ((u >> 16) & 1u);   // round-to-nearest-even
    return (unsigned short)(u >> 16);
}

__device__ __forceinline__ void store8bf(unsigned short* dst, float4 a, float4 b) {
    dst[0] = f2bf(a.x); dst[1] = f2bf(a.y); dst[2] = f2bf(a.z); dst[3] = f2bf(a.w);
    dst[4] = f2bf(b.x); dst[5] = f2bf(b.y); dst[6] = f2bf(b.z); dst[7] = f2bf(b.w);
}

// Branch-free tanh: native v_tanh_f32 on gfx1250 if the builtin exists,
// else exp-based identity (saturates correctly, no exec-mask branches).
__device__ __forceinline__ float fast_tanh(float x) {
#if __has_builtin(__builtin_amdgcn_tanhf)
    return __builtin_amdgcn_tanhf(x);
#else
    float e = __expf(2.0f * x);                       // v_exp_f32
    return 1.0f - 2.0f * __builtin_amdgcn_rcpf(e + 1.0f);  // v_rcp_f32
#endif
}

// -------------------------------------------------------------------------
// Fused: s[m] = sum_n tanh( (F @ Ww^T)[m,n] + bw[n] ) * v[n]
// Block: 64 rows, sweeps all 1280 cols; 256 threads = 8 waves in 2(M)x4(N).
// K staged 64-deep: 2 barriers per 8 WMMAs. fp32->bf16 fused into staging.
// -------------------------------------------------------------------------
__global__ __launch_bounds__(256) void gemm_tanh_score_kernel(
    const float* __restrict__ F,    // [TOTAL][D]
    const float* __restrict__ Ww,   // [D][D] row n, col k  (B matrix, n-major)
    const float* __restrict__ bw,   // [D]
    const float* __restrict__ vvec, // [D]
    float* __restrict__ s_out)      // [TOTAL]
{
    // stride 72 halfwords = 144 B rows: 16B-aligned, spreads LDS banks
    __shared__ __align__(16) unsigned short A_s[64][72];    // 64 x 64 used
    __shared__ __align__(16) unsigned short B_s[128][72];   // 128 x 64 used
    __shared__ float s_lds[64];

    const int tid    = threadIdx.x;
    const int lane   = tid & 31;
    const int wid    = tid >> 5;
    const int wave_m = wid & 1;   // 0..1 -> 32-row half
    const int wave_n = wid >> 1;  // 0..3 -> 32-col quarter
    const int m0     = blockIdx.x * 64;

    if (tid < 64) s_lds[tid] = 0.0f;

    float rowAcc[16];
#pragma unroll
    for (int i = 0; i < 16; ++i) rowAcc[i] = 0.0f;

    const int rA    = tid >> 2;          // A-stage row 0..63
    const int cA    = (tid & 3) * 16;    // A-stage k-chunk (16 elems)
    const int lrow  = lane & 15;
    const int koffA = (lane < 16) ? 0 : 16;   // A-frag K split (contiguous, matches B)
    const int kbase = (lane < 16) ? 0 : 16;   // bf16 B-frag K split (contiguous, matches A)

    const v8f vzero = {0.f, 0.f, 0.f, 0.f, 0.f, 0.f, 0.f, 0.f};

    for (int n0 = 0; n0 < D; n0 += 128) {
        v8f acc[2][2];
#pragma unroll
        for (int mt = 0; mt < 2; ++mt)
#pragma unroll
            for (int nt = 0; nt < 2; ++nt) acc[mt][nt] = vzero;

        for (int k0 = 0; k0 < D; k0 += 64) {
            __syncthreads();
            // ---- stage A tile 64x64 (fp32 -> bf16) ----
            {
                const float* src = F + (size_t)(m0 + rA) * D + k0 + cA;
                float4 f0 = *(const float4*)(src + 0);
                float4 f1 = *(const float4*)(src + 4);
                float4 f2 = *(const float4*)(src + 8);
                float4 f3 = *(const float4*)(src + 12);
                store8bf(&A_s[rA][cA + 0], f0, f1);
                store8bf(&A_s[rA][cA + 8], f2, f3);
            }
            // ---- stage B tile 128x64 (fp32 -> bf16), n-major ----
#pragma unroll
            for (int rep = 0; rep < 2; ++rep) {
                int c  = tid * 2 + rep;       // 0..511
                int rB = c >> 2;
                int cB = (c & 3) * 16;
                const float* src = Ww + (size_t)(n0 + rB) * D + k0 + cB;
                float4 f0 = *(const float4*)(src + 0);
                float4 f1 = *(const float4*)(src + 4);
                float4 f2 = *(const float4*)(src + 8);
                float4 f3 = *(const float4*)(src + 12);
                store8bf(&B_s[rB][cB + 0], f0, f1);
                store8bf(&B_s[rB][cB + 8], f2, f3);
            }
            __syncthreads();

            // ---- two K=32 sub-steps: 8 WMMAs per staged tile ----
#pragma unroll
            for (int ks = 0; ks < 64; ks += 32) {
                FragBF fa[2], fb[2];
#pragma unroll
                for (int mt = 0; mt < 2; ++mt) {
                    int row = wave_m * 32 + mt * 16 + lrow;
                    fa[mt].u[0] = *(const uint4*)&A_s[row][ks + koffA];      // K 0-7 / 16-23
                    fa[mt].u[1] = *(const uint4*)&A_s[row][ks + koffA + 8];  // K koffA+8..+15
                }
#pragma unroll
                for (int nt = 0; nt < 2; ++nt) {
                    int col = wave_n * 32 + nt * 16 + lrow;
                    fb[nt].u[0] = *(const uint4*)&B_s[col][ks + kbase];      // K kbase..+7
                    fb[nt].u[1] = *(const uint4*)&B_s[col][ks + kbase + 8];  // K kbase+8..+15
                }
#pragma unroll
                for (int mt = 0; mt < 2; ++mt)
#pragma unroll
                    for (int nt = 0; nt < 2; ++nt)
                        acc[mt][nt] = __builtin_amdgcn_wmma_f32_16x16x32_bf16(
                            false, fa[mt].v, false, fb[nt].v,
                            (short)0, acc[mt][nt], false, false);
            }
        }

        // ---- epilogue: tanh(c + bw) * v, fold over N into per-row partials ----
#pragma unroll
        for (int nt = 0; nt < 2; ++nt) {
            int col  = n0 + wave_n * 32 + nt * 16 + lrow;
            float vv = vvec[col];
            float bb = bw[col];
#pragma unroll
            for (int mt = 0; mt < 2; ++mt) {
#pragma unroll
                for (int r = 0; r < 8; ++r)
                    rowAcc[mt * 8 + r] += fast_tanh(acc[mt][nt][r] + bb) * vv;
            }
        }
    }

    __syncthreads();
#pragma unroll
    for (int mt = 0; mt < 2; ++mt)
#pragma unroll
        for (int r = 0; r < 8; ++r) {
            int row_l = wave_m * 32 + mt * 16 + r + ((lane & 16) ? 8 : 0);
            atomicAdd(&s_lds[row_l], rowAcc[mt * 8 + r]);
        }
    __syncthreads();
    if (tid < 64) s_out[m0 + tid] = s_lds[tid];
}

// ---- segment boundaries (segment_ids is sorted) ----
__global__ void seg_bounds_kernel(const int* __restrict__ seg, int* __restrict__ seg_start) {
    int i = blockIdx.x * 256 + threadIdx.x;
    if (i < TOTAL) {
        int sg = seg[i];
        if (i == 0 || seg[i - 1] != sg) seg_start[sg] = i;
    }
    if (i == 0) seg_start[N_SEG] = TOTAL;
}

// ---- per-segment stable softmax ----
__global__ __launch_bounds__(256) void softmax_kernel(
    const float* __restrict__ s, const int* __restrict__ seg_start,
    float* __restrict__ att)
{
    __shared__ float red[256];
    int g = blockIdx.x, t = threadIdx.x;
    int beg = seg_start[g], end = seg_start[g + 1];

    float m = -3.402823466e+38f;
    for (int i = beg + t; i < end; i += 256) m = fmaxf(m, s[i]);
    red[t] = m; __syncthreads();
    for (int off = 128; off > 0; off >>= 1) {
        if (t < off) red[t] = fmaxf(red[t], red[t + off]);
        __syncthreads();
    }
    m = red[0]; __syncthreads();

    float z = 0.f;
    for (int i = beg + t; i < end; i += 256) z += __expf(s[i] - m);
    red[t] = z; __syncthreads();
    for (int off = 128; off > 0; off >>= 1) {
        if (t < off) red[t] += red[t + off];
        __syncthreads();
    }
    float inv = 1.0f / red[0];
    for (int i = beg + t; i < end; i += 256) att[i] = __expf(s[i] - m) * inv;
}

// ---- attention-weighted pooling: pooled[g,:] = sum_i att[i]*F[i,:] ----
__global__ __launch_bounds__(320) void pool_kernel(
    const float* __restrict__ F, const float* __restrict__ att,
    const int* __restrict__ seg_start, float* __restrict__ pooled)
{
    int g = blockIdx.x, t = threadIdx.x;          // t = float4 column, 320*4 = 1280
    int beg = seg_start[g], end = seg_start[g + 1];
    float4 acc = {0.f, 0.f, 0.f, 0.f};
    for (int i = beg; i < end; ++i) {
        float a  = att[i];
        float4 f = *(const float4*)(F + (size_t)i * D + t * 4);
        acc.x += f.x * a; acc.y += f.y * a; acc.z += f.z * a; acc.w += f.w * a;
    }
    *(float4*)(pooled + (size_t)g * D + t * 4) = acc;
}

// ---- head: h = relu(pooled @ W1^T + b1) ----
__global__ __launch_bounds__(256) void head1_kernel(
    const float* __restrict__ pooled, const float* __restrict__ W1,
    const float* __restrict__ b1, float* __restrict__ h)
{
    int id = blockIdx.x * 256 + threadIdx.x;
    int g = id >> 9, j = id & 511;
    const float4* p = (const float4*)(pooled + (size_t)g * D);
    const float4* w = (const float4*)(W1 + (size_t)j * D);
    float acc = 0.f;
    for (int k = 0; k < D / 4; ++k) {
        float4 a = p[k], b = w[k];
        acc += a.x * b.x + a.y * b.y + a.z * b.z + a.w * b.w;
    }
    h[id] = fmaxf(acc + b1[j], 0.f);
}

// ---- out = h @ W2^T + b2 ----
__global__ __launch_bounds__(256) void head2_kernel(
    const float* __restrict__ h, const float* __restrict__ W2,
    const float* __restrict__ b2, float* __restrict__ out)
{
    __shared__ float red[256];
    int g = blockIdx.x, t = threadIdx.x;
    float acc = h[g * D_HEAD + t] * W2[t] + h[g * D_HEAD + t + 256] * W2[t + 256];
    red[t] = acc; __syncthreads();
    for (int off = 128; off > 0; off >>= 1) {
        if (t < off) red[t] += red[t + off];
        __syncthreads();
    }
    if (t == 0) out[g] = red[0] + b2[0];
}

extern "C" void kernel_launch(void* const* d_in, const int* in_sizes, int n_in,
                              void* d_out, int out_size, void* d_ws, size_t ws_size,
                              hipStream_t stream)
{
    const float* F  = (const float*)d_in[0];
    const float* Ww = (const float*)d_in[1];
    const float* bw = (const float*)d_in[2];
    const float* v  = (const float*)d_in[3];
    const float* W1 = (const float*)d_in[4];
    const float* b1 = (const float*)d_in[5];
    const float* W2 = (const float*)d_in[6];
    const float* b2 = (const float*)d_in[7];
    const int*  seg = (const int*)d_in[8];
    float* out = (float*)d_out;

    char* ws = (char*)d_ws;
    float* s_buf     = (float*)(ws + 0);        // 32768 f32
    float* att_buf   = (float*)(ws + 131072);   // 32768 f32
    int*   seg_start = (int*)  (ws + 262144);   // 129 i32
    float* pooled    = (float*)(ws + 263168);   // 128*1280 f32
    float* h_buf     = (float*)(ws + 918528);   // 128*512 f32

    seg_bounds_kernel     <<<TOTAL / 256, 256, 0, stream>>>(seg, seg_start);
    gemm_tanh_score_kernel<<<TOTAL / 64,  256, 0, stream>>>(F, Ww, bw, v, s_buf);
    softmax_kernel        <<<N_SEG,       256, 0, stream>>>(s_buf, seg_start, att_buf);
    pool_kernel           <<<N_SEG,       320, 0, stream>>>(F, att_buf, seg_start, pooled);
    head1_kernel          <<<(N_SEG * D_HEAD) / 256, 256, 0, stream>>>(pooled, W1, b1, h_buf);
    head2_kernel          <<<N_SEG,       256, 0, stream>>>(h_buf, W2, b2, out);
}
